// GIN_pyg_82094004896163
// MI455X (gfx1250) — matrix operand order repaired and checked by reading, not compiled, not run on hardware
//
#include <hip/hip_runtime.h>
#include <hip/hip_bf16.h>

#define DCH 128   // hidden channels (fixed by the model)

typedef __attribute__((ext_vector_type(16))) __bf16 v16bf;
typedef __attribute__((ext_vector_type(8)))  __bf16 v8bf;
typedef __attribute__((ext_vector_type(4)))  __bf16 v4bf;
typedef __attribute__((ext_vector_type(8)))  float  v8f;

// ---------------- utility kernels ----------------

__global__ void fill_zero_f32(float* __restrict__ p, int n) {
  int i = blockIdx.x * blockDim.x + threadIdx.x;
  if (i < n) p[i] = 0.0f;
}

__global__ void deg_accum(const int* __restrict__ dst, float* __restrict__ deg, int e) {
  int i = blockIdx.x * blockDim.x + threadIdx.x;
  if (i < e) atomicAdd(&deg[dst[i]], 1.0f);
}

__global__ void deg_to_dis(float* __restrict__ d, int n) {
  int i = blockIdx.x * blockDim.x + threadIdx.x;
  if (i < n) d[i] = rsqrtf(d[i] + 1.0f);   // deg^-1/2 with self-loop
}

// float4 -> 4x bf16 per thread
__global__ void f32_to_bf16(const float* __restrict__ x, __bf16* __restrict__ y, int n4) {
  int i = blockIdx.x * blockDim.x + threadIdx.x;
  if (i < n4) {
    float4 v = *reinterpret_cast<const float4*>(x + (size_t)i * 4);
    v4bf o = { (__bf16)v.x, (__bf16)v.y, (__bf16)v.z, (__bf16)v.w };
    *reinterpret_cast<v4bf*>(y + (size_t)i * 4) = o;
  }
}

// Wt[n][k] = bf16(W[k][n])  (128x128, one-shot, tiny)
__global__ void transpose_w_bf16(const float* __restrict__ W, __bf16* __restrict__ Wt) {
  int i = blockIdx.x * blockDim.x + threadIdx.x;   // 0 .. 128*128-1
  int n = i >> 7, k = i & (DCH - 1);
  Wt[n * DCH + k] = (__bf16)W[k * DCH + n];
}

// ---------------- WMMA GEMM: C[N,128] = A[N,128](bf16) @ W ; Wt = W^T bf16 ----------------
// One wave: 1 row tile x 4 col tiles (A registers reused 4x) -> 16 wmma/wave.
__global__ void gemm_wmma_bf16(const __bf16* __restrict__ A,
                               const __bf16* __restrict__ Wt,
                               float* __restrict__ C, int nrows) {
  const int lane = threadIdx.x & 31;
  const int wave = threadIdx.x >> 5;
  const int tm   = blockIdx.x * 8 + wave;          // row tile
  if (tm * 16 >= nrows) return;                    // wave-uniform: EXEC stays all-ones
  const int tn0  = blockIdx.y * 4;                 // first of 4 col tiles
  const int half = lane >> 4;                      // lanes 0-15 vs 16-31
  const int l16  = lane & 15;
  const bool fulltile = (tm * 16 + 16) <= nrows;   // wave-uniform

  int row = tm * 16 + l16;
  if (row >= nrows) row = nrows - 1;               // clamp loads; stores masked in slow path

  const __bf16* __restrict__ arow = A  + (size_t)row * DCH;
  const __bf16* __restrict__ bp0  = Wt + (size_t)(tn0 * 16 + l16) * DCH;

  v8f acc0 = {}, acc1 = {}, acc2 = {}, acc3 = {};
#pragma unroll
  for (int kb = 0; kb < 4; ++kb) {
    const int k0 = kb * 32;
    // A operand, contiguous k order used here (B matches): lanes 0-15 hold K=0-15;
    // lanes 16-31 hold K=16-31.
    v8bf alo = *reinterpret_cast<const v8bf*>(arow + k0 + half * 16);
    v8bf ahi = *reinterpret_cast<const v8bf*>(arow + k0 + half * 16 + 8);
    v16bf a;
#pragma unroll
    for (int i = 0; i < 8; ++i) { a[i] = alo[i]; a[i + 8] = ahi[i]; }
    // B operand, same contiguous k order as A: lane -> column, lanes 0-15 K=0-15, lanes 16-31 K=16-31.
    v16bf b0 = *reinterpret_cast<const v16bf*>(bp0 + 0 * 16 * DCH + k0 + half * 16);
    v16bf b1 = *reinterpret_cast<const v16bf*>(bp0 + 1 * 16 * DCH + k0 + half * 16);
    v16bf b2 = *reinterpret_cast<const v16bf*>(bp0 + 2 * 16 * DCH + k0 + half * 16);
    v16bf b3 = *reinterpret_cast<const v16bf*>(bp0 + 3 * 16 * DCH + k0 + half * 16);
    acc0 = __builtin_amdgcn_wmma_f32_16x16x32_bf16(false, a, false, b0, (short)0, acc0, false, false);
    acc1 = __builtin_amdgcn_wmma_f32_16x16x32_bf16(false, a, false, b1, (short)0, acc1, false, false);
    acc2 = __builtin_amdgcn_wmma_f32_16x16x32_bf16(false, a, false, b2, (short)0, acc2, false, false);
    acc3 = __builtin_amdgcn_wmma_f32_16x16x32_bf16(false, a, false, b3, (short)0, acc3, false, false);
  }

  // f32 C/D layout: VGPR v -> row (v + 8*half) within the tile, col = lane&15.
  if (fulltile) {
    float* __restrict__ crow = C + (size_t)(tm * 16 + 8 * half) * DCH + tn0 * 16 + l16;
#pragma unroll
    for (int v = 0; v < 8; ++v) {
      float* cp = crow + (size_t)v * DCH;
      cp[0]  = acc0[v];
      cp[16] = acc1[v];
      cp[32] = acc2[v];
      cp[48] = acc3[v];
    }
  } else {
#pragma unroll
    for (int v = 0; v < 8; ++v) {
      int r = tm * 16 + v + 8 * half;
      if (r < nrows) {
        float* cp = C + (size_t)r * DCH + tn0 * 16 + l16;
        cp[0]  = acc0[v];
        cp[16] = acc1[v];
        cp[32] = acc2[v];
        cp[48] = acc3[v];
      }
    }
  }
}

// ---------------- edge message scatter (roofline-dominant phase) ----------------
// 32 threads per edge, 4 channels/thread: one b128 gather + 4 f32 atomics.
__global__ void edge_scatter(const int* __restrict__ src, const int* __restrict__ dst,
                             const float* __restrict__ dis, const float* __restrict__ h,
                             float* __restrict__ agg, int e) {
  int gid = blockIdx.x * blockDim.x + threadIdx.x;
  int ed = gid >> 5;
  int c4 = (gid & 31) * 4;
  if (ed < e) {
    int s = src[ed], d = dst[ed];
    float coef = dis[s] * dis[d];
    float4 hv = *reinterpret_cast<const float4*>(h + (size_t)s * DCH + c4);
    float* ap = agg + (size_t)d * DCH + c4;
    atomicAdd(ap + 0, hv.x * coef);
    atomicAdd(ap + 1, hv.y * coef);
    atomicAdd(ap + 2, hv.z * coef);
    atomicAdd(ap + 3, hv.w * coef);
  }
}

// out = relu(agg + h*dis^2 + bias [+ residual]); float4 granularity.
// Writes f32 (residual/pooling) and bf16 (next layer's GEMM operand).
__global__ void combine_relu(const float* __restrict__ agg, const float* __restrict__ h,
                             const float* __restrict__ dis, const float* __restrict__ bias,
                             float* __restrict__ xf, __bf16* __restrict__ xb,
                             int n4, int use_resid) {
  int i = blockIdx.x * blockDim.x + threadIdx.x;
  if (i < n4) {
    int base = i * 4;
    int node = base >> 7;
    int c = base & (DCH - 1);
    float ds = dis[node];
    float dsq = ds * ds;
    float4 av = *reinterpret_cast<const float4*>(agg + base);
    float4 hv = *reinterpret_cast<const float4*>(h + base);
    float4 bv = *reinterpret_cast<const float4*>(bias + c);
    float4 v;
    v.x = fmaf(hv.x, dsq, av.x) + bv.x;
    v.y = fmaf(hv.y, dsq, av.y) + bv.y;
    v.z = fmaf(hv.z, dsq, av.z) + bv.z;
    v.w = fmaf(hv.w, dsq, av.w) + bv.w;
    if (use_resid) {
      float4 rv = *reinterpret_cast<const float4*>(xf + base);
      v.x += rv.x; v.y += rv.y; v.z += rv.z; v.w += rv.w;
    }
    v.x = fmaxf(v.x, 0.0f); v.y = fmaxf(v.y, 0.0f);
    v.z = fmaxf(v.z, 0.0f); v.w = fmaxf(v.w, 0.0f);
    *reinterpret_cast<float4*>(xf + base) = v;
    v4bf o = { (__bf16)v.x, (__bf16)v.y, (__bf16)v.z, (__bf16)v.w };
    *reinterpret_cast<v4bf*>(xb + base) = o;
  }
}

// ---------------- pooling + head ----------------

__global__ void pool_count(const int* __restrict__ batch, float* __restrict__ counts, int n) {
  int i = blockIdx.x * blockDim.x + threadIdx.x;
  if (i < n) atomicAdd(&counts[batch[i]], 1.0f);
}

// 32 threads per node, 4 channels/thread
__global__ void pool_sum(const float* __restrict__ xf, const int* __restrict__ batch,
                         float* __restrict__ pooled, int n) {
  int gid = blockIdx.x * blockDim.x + threadIdx.x;
  int node = gid >> 5;
  int c4 = (gid & 31) * 4;
  if (node < n) {
    int g = batch[node];
    float4 v = *reinterpret_cast<const float4*>(xf + (size_t)node * DCH + c4);
    float* pp = pooled + (size_t)g * DCH + c4;
    atomicAdd(pp + 0, v.x);
    atomicAdd(pp + 1, v.y);
    atomicAdd(pp + 2, v.z);
    atomicAdd(pp + 3, v.w);
  }
}

__global__ void head_kernel(const float* __restrict__ pooled, const float* __restrict__ counts,
                            const float* __restrict__ Wp, const float* __restrict__ bp,
                            float* __restrict__ out, int ng) {
  int i = blockIdx.x * blockDim.x + threadIdx.x;   // ng*2 threads
  if (i < ng * 2) {
    int g = i >> 1, o = i & 1;
    float inv = 1.0f / fmaxf(counts[g], 1.0f);
    float s = 0.0f;
#pragma unroll 4
    for (int c = 0; c < DCH; ++c) s += pooled[g * DCH + c] * Wp[c * 2 + o];
    out[i] = s * inv + bp[o];
  }
}

// ---------------- driver ----------------

extern "C" void kernel_launch(void* const* d_in, const int* in_sizes, int n_in,
                              void* d_out, int out_size, void* d_ws, size_t ws_size,
                              hipStream_t stream) {
  const float* x    = (const float*)d_in[0];
  const int*   ei   = (const int*)d_in[1];
  const int*   batch= (const int*)d_in[2];
  const float* W0   = (const float*)d_in[3];
  const float* b0   = (const float*)d_in[4];
  const float* Wh   = (const float*)d_in[5];
  const float* bh   = (const float*)d_in[6];
  const float* Wp   = (const float*)d_in[7];
  const float* bp   = (const float*)d_in[8];
  float* out = (float*)d_out;

  const int E  = in_sizes[1] / 2;
  const int N  = in_sizes[2];
  const int NG = out_size / 2;
  const int ND = N * DCH;

  const int* src = ei;
  const int* dst = ei + E;

  // carve workspace (256B aligned slices)
  char* w = (char*)d_ws;
  auto carve = [&](size_t bytes) -> char* {
    char* p = w;
    w += (bytes + 255) & ~(size_t)255;
    return p;
  };
  float*  dis    = (float*) carve((size_t)N * sizeof(float));
  float*  h      = (float*) carve((size_t)ND * sizeof(float));
  float*  agg    = (float*) carve((size_t)ND * sizeof(float));
  float*  xf     = (float*) carve((size_t)ND * sizeof(float));
  __bf16* xb     = (__bf16*)carve((size_t)ND * sizeof(__bf16));
  __bf16* Wt     = (__bf16*)carve((size_t)DCH * DCH * sizeof(__bf16));
  float*  pooled = (float*) carve((size_t)NG * DCH * sizeof(float));
  float*  counts = (float*) carve((size_t)NG * sizeof(float));

  const int T = 256;
  auto nb = [](long long n, int t) { return (unsigned)((n + t - 1) / t); };

  // symmetric-normalization coefficients: dis = (deg_dst + 1)^-1/2
  fill_zero_f32<<<nb(N, T), T, 0, stream>>>(dis, N);
  deg_accum<<<nb(E, T), T, 0, stream>>>(dst, dis, E);
  deg_to_dis<<<nb(N, T), T, 0, stream>>>(dis, N);

  // layer-0 GEMM operand
  f32_to_bf16<<<nb(ND / 4, T), T, 0, stream>>>(x, xb, ND / 4);

  const int row_tiles = (N + 15) / 16;
  dim3 ggrid(nb(row_tiles, 8), (DCH / 16) / 4);   // 8 waves/block (rows), 2 col-tile groups

  for (int layer = 0; layer < 4; ++layer) {
    const float* Wl = (layer == 0) ? W0 : (Wh + (size_t)(layer - 1) * DCH * DCH);
    const float* bl = (layer == 0) ? b0 : (bh + (size_t)(layer - 1) * DCH);
    transpose_w_bf16<<<nb(DCH * DCH, T), T, 0, stream>>>(Wl, Wt);
    gemm_wmma_bf16<<<ggrid, 256, 0, stream>>>(xb, Wt, h, N);
    fill_zero_f32<<<nb(ND, T), T, 0, stream>>>(agg, ND);
    edge_scatter<<<nb((long long)E * 32, T), T, 0, stream>>>(src, dst, dis, h, agg, E);
    combine_relu<<<nb(ND / 4, T), T, 0, stream>>>(agg, h, dis, bl, xf, xb, ND / 4, layer > 0);
  }

  // global mean pool + linear head
  fill_zero_f32<<<nb(NG * DCH, T), T, 0, stream>>>(pooled, NG * DCH);
  fill_zero_f32<<<1, T, 0, stream>>>(counts, NG);
  pool_count<<<nb(N, T), T, 0, stream>>>(batch, counts, N);
  pool_sum<<<nb((long long)N * 32, T), T, 0, stream>>>(xf, batch, pooled, N);
  head_kernel<<<nb((long long)NG * 2, T), T, 0, stream>>>(pooled, counts, Wp, bp, out, NG);

  (void)n_in; (void)ws_size;
}
